// SpikingCoreFlow_13950053778133
// MI455X (gfx1250) — matrix operand order repaired and checked by reading, not compiled, not run on hardware
//
#include <hip/hip_runtime.h>

typedef __attribute__((ext_vector_type(16))) _Float16 v16h;
typedef __attribute__((ext_vector_type(8)))  _Float16 v8h;
typedef __attribute__((ext_vector_type(8)))  float    v8f;

#define LNUM   4
#define BATCH  32
#define HDIM   1024
#define TSTEPS 128
#define NCYCLES (TSTEPS + LNUM)   // 132

// workspace layout (in floats)
#define WS_XI    0
#define WS_MEMB  (WS_XI   + BATCH*HDIM)          // L*B*H membranes
#define WS_BUFA  (WS_MEMB + LNUM*BATCH*HDIM)     // spike buffers (ping)
#define WS_BUFB  (WS_BUFA + LNUM*BATCH*HDIM)     // spike buffers (pong)
#define WS_ACC3  (WS_BUFB + LNUM*BATCH*HDIM)     // layer-3 spike accumulator B*H
#define WS_TOTAL (WS_ACC3 + BATCH*HDIM)          // ~458K floats = 1.75 MB
// optional f16 copy of W appended after the float region (8 MB)
#define WH_ELEMS ((size_t)LNUM * HDIM * HDIM)
#define WS_BYTES_BASE ((size_t)WS_TOTAL * sizeof(float))
#define WS_BYTES_WH   (WS_BYTES_BASE + WH_ELEMS * sizeof(_Float16))

// ---------------------------------------------------------------------------
// Init: zero all state, precompute xi = rint(x*T)
// ---------------------------------------------------------------------------
__global__ void snn_init_kernel(const float* __restrict__ x, float* __restrict__ ws) {
    const int stride = gridDim.x * blockDim.x;
    const int tid = blockIdx.x * blockDim.x + threadIdx.x;
    const int nzero = WS_TOTAL - WS_MEMB;
    for (int i = tid; i < nzero; i += stride)
        ws[WS_MEMB + i] = 0.0f;
    for (int i = tid; i < BATCH * HDIM; i += stride)
        ws[WS_XI + i] = rintf(x[i] * (float)TSTEPS);   // round-half-even matches jnp.round
}

// ---------------------------------------------------------------------------
// One-time fp32 -> f16 conversion of W (loop-invariant across all cycles):
// halves the hot loop's L2 bandwidth and removes per-K-step cvts.
// ---------------------------------------------------------------------------
__global__ void snn_wcvt_kernel(const float* __restrict__ W, _Float16* __restrict__ wh) {
    const size_t stride = (size_t)gridDim.x * blockDim.x * 8;
    for (size_t i = ((size_t)blockIdx.x * blockDim.x + threadIdx.x) * 8;
         i < WH_ELEMS; i += stride) {
        const v8f wv = *(const v8f*)(W + i);
        v8h hv;
        #pragma unroll
        for (int e = 0; e < 8; e++) hv[e] = (_Float16)wv[e];
        *(v8h*)(wh + i) = hv;
    }
}

// ---------------------------------------------------------------------------
// One simulation cycle. grid = LNUM*16 workgroups (16 N-strips of 64 per
// layer), 128 threads = 4 wave32; each wave owns a 16-wide N strip, M=32 via
// two 16x16 accumulators. sig (32x1024 f16) staged in LDS.
// WH=true: B-fragments from pre-converted f16 W; WH=false: inline fp32->f16.
// ---------------------------------------------------------------------------
template <bool WH>
__launch_bounds__(128)
__global__ void snn_step_kernel(const float*    __restrict__ W,
                                const _Float16* __restrict__ wh,
                                const float*    __restrict__ thresholds,
                                const int*      __restrict__ src_idx,
                                float*          __restrict__ ws,
                                const float*    __restrict__ bufold,
                                float*          __restrict__ bufnew,
                                int cycle) {
    __shared__ _Float16 sig[BATCH * HDIM];    // 64 KB

    const int l   = blockIdx.x >> 4;          // layer
    const int nwg = (blockIdx.x & 15) * 64;   // workgroup's output-column base
    const int tid = threadIdx.x;

    float*       memb = ws + WS_MEMB + l * BATCH * HDIM;
    float*       acc3 = ws + WS_ACC3;
    const float* bold = bufold + l * BATCH * HDIM;
    float*       bnew = bufnew + l * BATCH * HDIM;

    const bool active = (cycle >= l) && (cycle < TSTEPS + l);
    if (!active) {
        // masked-out layer: carry spike buffer through, membrane untouched
        for (int t = tid; t < BATCH * 64; t += 128) {
            const int b = t >> 6;
            const int o = nwg + (t & 63);
            bnew[b * HDIM + o] = bold[b * HDIM + o];
        }
        return;
    }

    // ---- gather phase: sig[b][i] = prev[b][src_idx[l][i]] as f16 ----------
    const float* prev = (l == 0) ? (ws + WS_XI) : (bufold + (l - 1) * BATCH * HDIM);
    const int*   sidx = src_idx + l * HDIM;
    {
        const int i0 = tid * 8;               // each thread owns 8 columns, all b
        int s[8];
        #pragma unroll
        for (int e = 0; e < 8; e++) s[e] = sidx[i0 + e];
        if (l == 0) {
            const float c = (float)cycle;
            for (int b = 0; b < BATCH; b++) {
                v8h pk;
                #pragma unroll
                for (int e = 0; e < 8; e++)
                    pk[e] = (prev[b * HDIM + s[e]] > c) ? (_Float16)1.0f : (_Float16)0.0f;
                *(v8h*)(&sig[b * HDIM + i0]) = pk;
            }
        } else {
            for (int b = 0; b < BATCH; b++) {
                v8h pk;
                #pragma unroll
                for (int e = 0; e < 8; e++)
                    pk[e] = (_Float16)prev[b * HDIM + s[e]];   // exact: 0.0 or 1.0
                *(v8h*)(&sig[b * HDIM + i0]) = pk;
            }
        }
    }
    __syncthreads();

    // ---- WMMA phase: D[m,n] = sum_k sig[m,k] * W[l,n,k] --------------------
    const int wave = tid >> 5;
    const int lane = tid & 31;
    const int half = lane >> 4;               // 0: lanes 0-15, 1: lanes 16-31
    const int lr   = lane & 15;
    const int n    = nwg + wave * 16 + lr;    // this lane's output column

    const float*    wrow  = W  + ((size_t)l * HDIM + n) * (size_t)HDIM;  // W[l, n, :]
    const _Float16* whrow = wh + ((size_t)l * HDIM + n) * (size_t)HDIM;

    v8f c0 = {};    // rows 0..15
    v8f c1 = {};    // rows 16..31
    for (int k0 = 0; k0 < HDIM; k0 += 32) {
        // B fragment: lane holds column n, K = half*16 + 0..15
        union { v16h v; struct { v8h lo, hi; } s; } bf;
        if constexpr (WH) {
            const v8h* wp = (const v8h*)(whrow + k0 + half * 16);
            bf.s.lo = wp[0];
            bf.s.hi = wp[1];
        } else {
            const v8f* wp = (const v8f*)(wrow + k0 + half * 16);
            const v8f w0 = wp[0];
            const v8f w1 = wp[1];
            #pragma unroll
            for (int e = 0; e < 8; e++) {
                bf.s.lo[e] = (_Float16)w0[e];
                bf.s.hi[e] = (_Float16)w1[e];
            }
        }
        // A fragments: lane row = m, elems 0..15 = K half*16+{0..15}
        union { v16h v; struct { v8h lo, hi; } s; } a0, a1;
        const int m0 = lr;
        const int m1 = 16 + lr;
        a0.s.lo = *(const v8h*)(&sig[m0 * HDIM + k0 + half * 16]);
        a0.s.hi = *(const v8h*)(&sig[m0 * HDIM + k0 + half * 16 + 8]);
        a1.s.lo = *(const v8h*)(&sig[m1 * HDIM + k0 + half * 16]);
        a1.s.hi = *(const v8h*)(&sig[m1 * HDIM + k0 + half * 16 + 8]);

        c0 = __builtin_amdgcn_wmma_f32_16x16x32_f16(false, a0.v, false, bf.v,
                                                    (short)0, c0, false, false);
        c1 = __builtin_amdgcn_wmma_f32_16x16x32_f16(false, a1.v, false, bf.v,
                                                    (short)0, c1, false, false);
    }

    // ---- epilogue: membrane update, fire, reset, layer-3 accumulation ------
    const float thr = thresholds[l];
    // weight-2 at cycle 130 folds in the stale re-add of inactive cycle 131
    const float wgt = (cycle == TSTEPS + LNUM - 2) ? 2.0f : 1.0f;
    #pragma unroll
    for (int j = 0; j < 8; j++) {
        {   // M-tile 0: rows 0..15 ; C/D layout: VGPR j -> M = half*8 + j
            const int b = half * 8 + j;
            const size_t off = (size_t)b * HDIM + n;
            const float mv = memb[off] + c0[j];
            const float sp = (mv > thr) ? 1.0f : 0.0f;
            memb[off] = mv - sp * thr;
            bnew[off] = sp;
            if (l == LNUM - 1) acc3[off] += sp * wgt;
        }
        {   // M-tile 1: rows 16..31
            const int b = 16 + half * 8 + j;
            const size_t off = (size_t)b * HDIM + n;
            const float mv = memb[off] + c1[j];
            const float sp = (mv > thr) ? 1.0f : 0.0f;
            memb[off] = mv - sp * thr;
            bnew[off] = sp;
            if (l == LNUM - 1) acc3[off] += sp * wgt;
        }
    }
}

// ---------------------------------------------------------------------------
// Final: out[b,h] = acc3[b, out_idx[h]]
// ---------------------------------------------------------------------------
__global__ void snn_out_gather_kernel(const int* __restrict__ out_idx,
                                      const float* __restrict__ acc3,
                                      float* __restrict__ out) {
    const int idx = blockIdx.x * blockDim.x + threadIdx.x;
    if (idx < BATCH * HDIM) {
        const int b = idx >> 10;
        const int h = idx & (HDIM - 1);
        out[idx] = acc3[b * HDIM + out_idx[h]];
    }
}

// ---------------------------------------------------------------------------
extern "C" void kernel_launch(void* const* d_in, const int* in_sizes, int n_in,
                              void* d_out, int out_size, void* d_ws, size_t ws_size,
                              hipStream_t stream) {
    const float* x          = (const float*)d_in[0];   // (B,H)
    const float* W          = (const float*)d_in[1];   // (L,H,H)
    const float* thresholds = (const float*)d_in[2];   // (L,)
    const int*   src_idx    = (const int*)  d_in[3];   // (L,H)
    const int*   out_idx    = (const int*)  d_in[4];   // (H,)
    float* ws  = (float*)d_ws;
    float* out = (float*)d_out;
    _Float16* wh = (_Float16*)((char*)d_ws + WS_BYTES_BASE);

    // ws_size is a harness constant -> this branch is deterministic
    const bool use_wh = (ws_size >= WS_BYTES_WH);

    snn_init_kernel<<<512, 256, 0, stream>>>(x, ws);
    if (use_wh)
        snn_wcvt_kernel<<<2048, 256, 0, stream>>>(W, wh);

    float* bufA = ws + WS_BUFA;
    float* bufB = ws + WS_BUFB;
    for (int cycle = 0; cycle < NCYCLES; cycle++) {
        const float* bo = (cycle & 1) ? bufB : bufA;
        float*       bn = (cycle & 1) ? bufA : bufB;
        if (use_wh)
            snn_step_kernel<true><<<LNUM * 16, 128, 0, stream>>>(
                W, wh, thresholds, src_idx, ws, bo, bn, cycle);
        else
            snn_step_kernel<false><<<LNUM * 16, 128, 0, stream>>>(
                W, wh, thresholds, src_idx, ws, bo, bn, cycle);
    }

    snn_out_gather_kernel<<<(BATCH * HDIM + 255) / 256, 256, 0, stream>>>(
        out_idx, ws + WS_ACC3, out);
}
